// NeuralPolarDecoder_22686017257998
// MI455X (gfx1250) — matrix operand order repaired and checked by reading, not compiled, not run on hardware
//
#include <hip/hip_runtime.h>
#include <hip/hip_bf16.h>
#include <cstdint>

// ---------------------------------------------------------------------------
// Neural polar decoder on MI455X (gfx1250, wave32).
//  - All MLPs on v_wmma_f32_16x16x32_f16 (f16 operands, fp32 accumulate).
//  - Weights pre-packed fragment-major (1KB per 32x16 B-fragment, lane-
//    contiguous 32B runs) -> conflict-free LDS reads & coalesced global reads.
//  - Per-workgroup weight blob (57.6KB) moved to LDS via the Tensor Data
//    Mover (tensor_load_to_lds + s_wait_tensorcnt) when available.
// ---------------------------------------------------------------------------

#define BB   32
#define NNN  4096
#define BNP  (BB * NNN)   // 131072 positions
#define NLEV 12

// Fragment-major weight blob layout (halves), contiguous in workspace & LDS:
#define WOFF_CW1  0        // cn_w1 : K=128 -> 16 frags * 512 = 8192 halves
#define WOFF_CW2  8192     // cn_w2 : K=64  ->  8 frags       = 4096 halves
#define WOFF_BW1  12288    // bn_w1 : K=192 -> 24 frags       = 12288 halves
#define WOFF_BW2  24576    // bn_w2 : K=64  ->  8 frags       = 4096 halves
#define WOFF_LEMB 28672    // label_emb (2x64 row-major)      = 128 halves
#define WBLOB_HALVES 28800 // total -> 57600 bytes
#define STAGE_HALVES (8 * 16 * 64)
#define LEVEL_LDS_BYTES ((WBLOB_HALVES + STAGE_HALVES) * 2)

typedef _Float16 half8  __attribute__((ext_vector_type(8)));
typedef _Float16 half16 __attribute__((ext_vector_type(16)));
typedef float    float8 __attribute__((ext_vector_type(8)));
typedef unsigned int uint32x4 __attribute__((ext_vector_type(4)));
typedef int      int32x4 __attribute__((ext_vector_type(4)));
typedef int      int32x8 __attribute__((ext_vector_type(8)));

#if defined(__has_builtin)
#if __has_builtin(__builtin_amdgcn_tensor_load_to_lds) && \
    __has_builtin(__builtin_amdgcn_s_wait_tensorcnt)
#define USE_TDM 1
#endif
#endif
#ifndef USE_TDM
#define USE_TDM 0
#endif

union H16u { half16 v; half8 h[2]; };

__device__ __forceinline__ float8 wmma16(half16 a, half16 b, float8 c) {
  // D = A(16x32 f16) * B(32x16 f16) + C(16x16 f32)
  return __builtin_amdgcn_wmma_f32_16x16x32_f16(false, a, false, b, (short)0, c,
                                                false, false);
}

// A fragment (16x32 f16), lane half hi holds k0+16*hi .. k0+16*hi+15 (contiguous k order, as B).
__device__ __forceinline__ half16 a_frag(const _Float16* row, int k0, int hi) {
  H16u u;
  u.h[0] = *(const half8*)(row + k0 + hi * 16);
  u.h[1] = *(const half8*)(row + k0 + hi * 16 + 8);
  return u.v;
}

// B fragment (32x16 f16) from fragment-major storage: frag fid is a 1KB block,
// lane L's 16 halves at halves [fid*512 + L*16 .. +15].
__device__ __forceinline__ half16 b_frag_fm(const _Float16* base, int fid,
                                            int lane) {
  const _Float16* p = base + fid * 512 + lane * 16;
  H16u u;
  u.h[0] = *(const half8*)(p);
  u.h[1] = *(const half8*)(p + 8);
  return u.v;
}

// ---------------------------------------------------------------------------
// Prep: pack weights (row-major (K,64) fp32) into fragment-major f16 blocks.
// Fragment fid = (k0/32)*4 + (n0/16); within: lane=(o/16)&31, j=o&15 maps to
// n = (fid&3)*16 + (lane&15), k = (fid>>2)*32 + (lane>>4)*16 + j.
// ---------------------------------------------------------------------------
__device__ __forceinline__ void write_fm(_Float16* dst, const float* w, int K,
                                         int kvalid, int tid, int nth) {
  int total = 64 * K;
  for (int o = tid; o < total; o += nth) {
    int fid = o >> 9, lane = (o >> 4) & 31, j = o & 15;
    int n = (fid & 3) * 16 + (lane & 15);
    int k = (fid >> 2) * 32 + (lane >> 4) * 16 + j;
    dst[o] = (_Float16)(k < kvalid ? w[k * 64 + n] : 0.0f);
  }
}

__global__ void __launch_bounds__(256) prep_kernel(
    const float* __restrict__ emb_w1, const float* __restrict__ emb_w2,
    const float* __restrict__ cn_w1,  const float* __restrict__ cn_w2,
    const float* __restrict__ bn_w1,  const float* __restrict__ bn_w2,
    const float* __restrict__ label_emb,
    _Float16* __restrict__ ew1t, _Float16* __restrict__ ew2t,
    _Float16* __restrict__ wblob) {
  int tid = blockIdx.x * blockDim.x + threadIdx.x;
  int nth = gridDim.x * blockDim.x;
  write_fm(ew1t, emb_w1, 32, 2, tid, nth);          // (2,64) zero-padded K->32
  write_fm(ew2t, emb_w2, 64, 64, tid, nth);
  write_fm(wblob + WOFF_CW1, cn_w1, 128, 128, tid, nth);
  write_fm(wblob + WOFF_CW2, cn_w2, 64, 64, tid, nth);
  write_fm(wblob + WOFF_BW1, bn_w1, 192, 192, tid, nth);
  write_fm(wblob + WOFF_BW2, bn_w2, 64, 64, tid, nth);
  for (int i = tid; i < 2 * 64; i += nth)
    wblob[WOFF_LEMB + i] = (_Float16)label_emb[i];
}

// ---------------------------------------------------------------------------
// Embedding: e = relu(y @ w1 + b1) @ w2 + b2. One wave = 16 positions.
// ---------------------------------------------------------------------------
__global__ void __launch_bounds__(256) emb_kernel(
    const int* __restrict__ x, const float* __restrict__ y,
    const float* __restrict__ emb_b1, const float* __restrict__ emb_b2,
    const _Float16* __restrict__ ew1t, const _Float16* __restrict__ ew2t,
    _Float16* __restrict__ e_out, unsigned char* __restrict__ v_out) {
  __shared__ __align__(32) _Float16 stage_all[STAGE_HALVES];
  const int lane = threadIdx.x & 31;
  const int wv   = threadIdx.x >> 5;
  _Float16* st = stage_all + wv * 1024;
  const int nl = lane & 15, hi = lane >> 4;
  const int pos = (blockIdx.x * 8 + wv) * 16 + nl;

  if (hi == 0) v_out[pos] = (unsigned char)(x[pos] & 1);

  // Layer 1: K=2 padded to 32 (only lanes 0-15, halves 0-1 carry data).
  half16 a;
#pragma unroll
  for (int j = 0; j < 16; j++) a[j] = (_Float16)0.0f;
  if (hi == 0) {
    a[0] = (_Float16)y[pos * 2 + 0];
    a[1] = (_Float16)y[pos * 2 + 1];
  }
  float8 acc[4];
#pragma unroll
  for (int t = 0; t < 4; t++) {
    float bv = emb_b1[t * 16 + nl];
#pragma unroll
    for (int r = 0; r < 8; r++) acc[t][r] = bv;
  }
#pragma unroll
  for (int t = 0; t < 4; t++)
    acc[t] = wmma16(a, b_frag_fm(ew1t, t, lane), acc[t]);

#pragma unroll
  for (int t = 0; t < 4; t++)
#pragma unroll
    for (int r = 0; r < 8; r++) {
      float v = acc[t][r];
      st[(r + hi * 8) * 64 + t * 16 + nl] = (_Float16)(v > 0.0f ? v : 0.0f);
    }

  // Layer 2: K=64.
#pragma unroll
  for (int t = 0; t < 4; t++) {
    float bv = emb_b2[t * 16 + nl];
#pragma unroll
    for (int r = 0; r < 8; r++) acc[t][r] = bv;
  }
  const _Float16* hrow = st + nl * 64;
#pragma unroll
  for (int k0 = 0; k0 < 64; k0 += 32) {
    half16 aa = a_frag(hrow, k0, hi);
#pragma unroll
    for (int t = 0; t < 4; t++)
      acc[t] = wmma16(aa, b_frag_fm(ew2t, (k0 >> 5) * 4 + t, lane), acc[t]);
  }
#pragma unroll
  for (int t = 0; t < 4; t++)
#pragma unroll
    for (int r = 0; r < 8; r++)
      st[(r + hi * 8) * 64 + t * 16 + nl] = (_Float16)acc[t][r];
  {
    const _Float16* srow = st + nl * 64 + hi * 32;
    _Float16* drow = e_out + (size_t)pos * 64 + hi * 32;
#pragma unroll
    for (int j = 0; j < 4; j++)
      *(half8*)(drow + j * 8) = *(const half8*)(srow + j * 8);
  }
}

// ---------------------------------------------------------------------------
// One butterfly level. Weights blob is staged to LDS (via TDM if available),
// then one wave = 16 pairs: e_left (CN, K=128) and e_right (BN, K=192).
// ---------------------------------------------------------------------------
__global__ void __launch_bounds__(256) level_kernel(
    const _Float16* __restrict__ e_src, const unsigned char* __restrict__ v_src,
    _Float16* __restrict__ e_dst, unsigned char* __restrict__ v_dst,
    const float* __restrict__ cn_b1, const float* __restrict__ cn_b2,
    const float* __restrict__ bn_b1, const float* __restrict__ bn_b2,
    const _Float16* __restrict__ wblob, int lsh /* L = 1<<lsh */) {
  extern __shared__ _Float16 smem[];
  _Float16* lds_w = smem;                        // WBLOB_HALVES
  const int lane = threadIdx.x & 31;
  const int wv   = threadIdx.x >> 5;
  _Float16* st = smem + WBLOB_HALVES + wv * 1024;
  const int nl = lane & 15, hi = lane >> 4;

  // ---- Stage weight blob (57600B) into LDS ----
#if USE_TDM
  if (wv == 0) {
    unsigned long long ga = (unsigned long long)(uintptr_t)wblob;
    unsigned lds_off = (unsigned)(uintptr_t)lds_w;   // low 32 bits = LDS addr
    const unsigned n8 = (WBLOB_HALVES * 2) / 8;      // 7200 8-byte elements
    uint32x4 g0;
    g0[0] = 1u;                                      // count=1, user mode
    g0[1] = lds_off;                                 // lds_addr
    g0[2] = (unsigned)(ga & 0xFFFFFFFFu);            // global_addr[31:0]
    g0[3] = (unsigned)((ga >> 32) & 0x01FFFFFFu) | (2u << 30);  // [56:32]|type=2
    int32x8 g1;
    g1[0] = (int)(3u << 16);                         // data_size=8B, mask=0
    g1[1] = (int)((n8 & 0xFFFFu) << 16);             // tensor_dim0[15:0]
    g1[2] = (int)((n8 >> 16) & 0xFFFFu) | (1 << 16); // dim0[31:16] | dim1=1
    g1[3] = (int)((n8 & 0xFFFFu) << 16);             // tile_dim0 = n8
    g1[4] = 0;                                       // tile_dim1/2 unused
    g1[5] = (int)n8;                                 // dim0_stride[31:0]
    g1[6] = (int)((n8 & 0xFFFFu) << 16);             // dim1_stride[15:0]
    g1[7] = 0;
    int32x4 z4 = {0, 0, 0, 0};
#if defined(__clang_major__) && __clang_major__ >= 23
    int32x8 z8 = {0, 0, 0, 0, 0, 0, 0, 0};
    __builtin_amdgcn_tensor_load_to_lds(g0, g1, z4, z4, z8, 0);
#else
    __builtin_amdgcn_tensor_load_to_lds(g0, g1, z4, z4, 0);
#endif
    __builtin_amdgcn_s_wait_tensorcnt((short)0);
  }
  __syncthreads();
#else
  for (int c = threadIdx.x; c < WBLOB_HALVES / 8; c += 256)
    *(half8*)(lds_w + c * 8) = *(const half8*)(wblob + c * 8);
  __syncthreads();
#endif
  const _Float16* lds_cw1 = lds_w + WOFF_CW1;
  const _Float16* lds_cw2 = lds_w + WOFF_CW2;
  const _Float16* lds_bw1 = lds_w + WOFF_BW1;
  const _Float16* lds_bw2 = lds_w + WOFF_BW2;

  // ---- Pair -> (batch, block, index-in-block); same on both lane halves ----
  const int pair  = (blockIdx.x * 8 + wv) * 16 + nl;
  const int b     = pair >> 11;                 // / (N/2)
  const int pidx  = pair & 2047;
  const int halfL = 1 << (lsh - 1);
  const int blk   = pidx >> (lsh - 1);
  const int ii    = pidx & (halfL - 1);
  const int base  = b * NNN + (blk << lsh);
  const int src_pos   = base + 2 * ii;          // e_odd row; e_even adjacent
  const int dst_left  = base + ii;
  const int dst_right = dst_left + halfL;

  const int v_odd = v_src[src_pos], v_even = v_src[src_pos + 1];
  const int vx = v_odd ^ v_even;
  if (hi == 0) {
    v_dst[dst_left]  = (unsigned char)vx;
    v_dst[dst_right] = (unsigned char)v_even;
  }
  const _Float16* arow = e_src + (size_t)src_pos * 64;  // 128 contiguous halves
  const _Float16* lrow = lds_w + WOFF_LEMB + vx * 64;   // label_emb[v_xor]

  float8 acc[4];

  // ---------------- CN layer 1: (16x128) @ (128x64) ----------------
#pragma unroll
  for (int t = 0; t < 4; t++) {
    float bv = cn_b1[t * 16 + nl];
#pragma unroll
    for (int r = 0; r < 8; r++) acc[t][r] = bv;
  }
#pragma unroll
  for (int k0 = 0; k0 < 128; k0 += 32) {
    half16 a = a_frag(arow, k0, hi);
#pragma unroll
    for (int t = 0; t < 4; t++)
      acc[t] = wmma16(a, b_frag_fm(lds_cw1, (k0 >> 5) * 4 + t, lane), acc[t]);
  }
#pragma unroll
  for (int t = 0; t < 4; t++)
#pragma unroll
    for (int r = 0; r < 8; r++) {
      float v = acc[t][r];
      st[(r + hi * 8) * 64 + t * 16 + nl] = (_Float16)(v > 0.0f ? v : 0.0f);
    }

  // ---------------- CN layer 2: (16x64) @ (64x64) ----------------
#pragma unroll
  for (int t = 0; t < 4; t++) {
    float bv = cn_b2[t * 16 + nl];
#pragma unroll
    for (int r = 0; r < 8; r++) acc[t][r] = bv;
  }
  {
    const _Float16* hrow = st + nl * 64;
#pragma unroll
    for (int k0 = 0; k0 < 64; k0 += 32) {
      half16 a = a_frag(hrow, k0, hi);
#pragma unroll
      for (int t = 0; t < 4; t++)
        acc[t] = wmma16(a, b_frag_fm(lds_cw2, (k0 >> 5) * 4 + t, lane), acc[t]);
    }
  }
#pragma unroll
  for (int t = 0; t < 4; t++)
#pragma unroll
    for (int r = 0; r < 8; r++)
      st[(r + hi * 8) * 64 + t * 16 + nl] = (_Float16)acc[t][r];
  {
    const _Float16* srow = st + nl * 64 + hi * 32;
    _Float16* drow = e_dst + (size_t)dst_left * 64 + hi * 32;
#pragma unroll
    for (int j = 0; j < 4; j++)
      *(half8*)(drow + j * 8) = *(const half8*)(srow + j * 8);
  }

  // ---------------- BN layer 1: (16x192) @ (192x64) ----------------
#pragma unroll
  for (int t = 0; t < 4; t++) {
    float bv = bn_b1[t * 16 + nl];
#pragma unroll
    for (int r = 0; r < 8; r++) acc[t][r] = bv;
  }
#pragma unroll
  for (int k0 = 0; k0 < 128; k0 += 32) {
    half16 a = a_frag(arow, k0, hi);
#pragma unroll
    for (int t = 0; t < 4; t++)
      acc[t] = wmma16(a, b_frag_fm(lds_bw1, (k0 >> 5) * 4 + t, lane), acc[t]);
  }
#pragma unroll
  for (int k0 = 128; k0 < 192; k0 += 32) {
    half16 a = a_frag(lrow, k0 - 128, hi);      // v_left = label_emb[v_xor]
#pragma unroll
    for (int t = 0; t < 4; t++)
      acc[t] = wmma16(a, b_frag_fm(lds_bw1, (k0 >> 5) * 4 + t, lane), acc[t]);
  }
#pragma unroll
  for (int t = 0; t < 4; t++)
#pragma unroll
    for (int r = 0; r < 8; r++) {
      float v = acc[t][r];
      st[(r + hi * 8) * 64 + t * 16 + nl] = (_Float16)(v > 0.0f ? v : 0.0f);
    }

  // ---------------- BN layer 2: (16x64) @ (64x64) ----------------
#pragma unroll
  for (int t = 0; t < 4; t++) {
    float bv = bn_b2[t * 16 + nl];
#pragma unroll
    for (int r = 0; r < 8; r++) acc[t][r] = bv;
  }
  {
    const _Float16* hrow = st + nl * 64;
#pragma unroll
    for (int k0 = 0; k0 < 64; k0 += 32) {
      half16 a = a_frag(hrow, k0, hi);
#pragma unroll
      for (int t = 0; t < 4; t++)
        acc[t] = wmma16(a, b_frag_fm(lds_bw2, (k0 >> 5) * 4 + t, lane), acc[t]);
    }
  }
#pragma unroll
  for (int t = 0; t < 4; t++)
#pragma unroll
    for (int r = 0; r < 8; r++)
      st[(r + hi * 8) * 64 + t * 16 + nl] = (_Float16)acc[t][r];
  {
    const _Float16* srow = st + nl * 64 + hi * 32;
    _Float16* drow = e_dst + (size_t)dst_right * 64 + hi * 32;
#pragma unroll
    for (int j = 0; j < 4; j++)
      *(half8*)(drow + j * 8) = *(const half8*)(srow + j * 8);
  }
}

// ---------------------------------------------------------------------------
// LLR head: p = softmax(e @ llr_w + llr_b); loss = -log(clip(p)[label]).
// ---------------------------------------------------------------------------
__global__ void __launch_bounds__(256) llr_kernel(
    const _Float16* __restrict__ e, const unsigned char* __restrict__ v,
    const float* __restrict__ llr_w, const float* __restrict__ llr_b,
    float* __restrict__ loss_out, float* __restrict__ pred1,
    float* __restrict__ pred2, int lvl) {
  const int pos = blockIdx.x * blockDim.x + threadIdx.x;   // 0..BNP-1
  const int b = pos >> 12, n = pos & (NNN - 1);
  const _Float16* er = e + (size_t)pos * 64;
  float l0 = llr_b[0], l1 = llr_b[1];
#pragma unroll
  for (int k8 = 0; k8 < 64; k8 += 8) {
    half8 ev = *(const half8*)(er + k8);
#pragma unroll
    for (int j = 0; j < 8; j++) {
      float ef = (float)ev[j];
      l0 += ef * llr_w[(k8 + j) * 2 + 0];
      l1 += ef * llr_w[(k8 + j) * 2 + 1];
    }
  }
  float m  = fmaxf(l0, l1);
  float x0 = expf(l0 - m), x1 = expf(l1 - m);
  float inv = 1.0f / (x0 + x1);
  float p0 = x0 * inv, p1 = x1 * inv;
  float c0 = fminf(fmaxf(p0, 1e-7f), 1.0f - 1e-7f);
  float c1 = fminf(fmaxf(p1, 1e-7f), 1.0f - 1e-7f);
  int lab = v[pos];
  float pl = lab ? c1 : c0;
  loss_out[((size_t)b * NLEV + lvl) * NNN + n] = -logf(pl);
  size_t po = (((size_t)b * NNN + n) * NLEV + lvl) * 2;
  pred1[po] = p0; pred1[po + 1] = p1;
  pred2[po] = p0; pred2[po + 1] = p1;
}

// ---------------------------------------------------------------------------
extern "C" void kernel_launch(void* const* d_in, const int* in_sizes, int n_in,
                              void* d_out, int out_size, void* d_ws,
                              size_t ws_size, hipStream_t stream) {
  const int*   x      = (const int*)  d_in[0];
  const float* y      = (const float*)d_in[1];
  const float* emb_w1 = (const float*)d_in[2];
  const float* emb_b1 = (const float*)d_in[3];
  const float* emb_w2 = (const float*)d_in[4];
  const float* emb_b2 = (const float*)d_in[5];
  const float* cn_w1  = (const float*)d_in[6];
  const float* cn_b1  = (const float*)d_in[7];
  const float* cn_w2  = (const float*)d_in[8];
  const float* cn_b2  = (const float*)d_in[9];
  const float* bn_w1  = (const float*)d_in[10];
  const float* bn_b1  = (const float*)d_in[11];
  const float* bn_w2  = (const float*)d_in[12];
  const float* bn_b2  = (const float*)d_in[13];
  const float* llr_w  = (const float*)d_in[14];
  const float* llr_b  = (const float*)d_in[15];
  const float* label_emb = (const float*)d_in[16];
  (void)in_sizes; (void)n_in; (void)out_size; (void)ws_size;

  // Workspace partition (256B aligned; blob pieces are 256B multiples so the
  // fragment-major weight blob stays contiguous for the TDM copy).
  char* ws = (char*)d_ws;
  size_t off = 0;
  auto walloc = [&](size_t bytes) {
    void* p = ws + off;
    off = (off + bytes + 255) & ~(size_t)255;
    return p;
  };
  _Float16* e0 = (_Float16*)walloc((size_t)BNP * 64 * 2);
  _Float16* e1 = (_Float16*)walloc((size_t)BNP * 64 * 2);
  unsigned char* v0 = (unsigned char*)walloc(BNP);
  unsigned char* v1 = (unsigned char*)walloc(BNP);
  _Float16* ew1t  = (_Float16*)walloc(64 * 32 * 2);
  _Float16* ew2t  = (_Float16*)walloc(64 * 64 * 2);
  _Float16* wblob = (_Float16*)walloc((size_t)WBLOB_HALVES * 2);

  float* loss  = (float*)d_out;                       // (B, 12, N)
  float* pred1 = loss + (size_t)BB * NLEV * NNN;      // (B, N, 12, 2)
  float* pred2 = pred1 + (size_t)BB * NNN * NLEV * 2; // duplicate output

  prep_kernel<<<32, 256, 0, stream>>>(emb_w1, emb_w2, cn_w1, cn_w2, bn_w1,
                                      bn_w2, label_emb, ew1t, ew2t, wblob);
  emb_kernel<<<BNP / 128, 256, 0, stream>>>(x, y, emb_b1, emb_b2, ew1t, ew2t,
                                            e0, v0);

  const _Float16* es = e0;  _Float16* ed = e1;
  const unsigned char* vs = v0;  unsigned char* vd = v1;
  for (int lvl = 0; lvl < NLEV; lvl++) {
    int lsh = NLEV - lvl;                    // block length L = 1 << lsh
    level_kernel<<<(BNP / 2) / 128, 256, LEVEL_LDS_BYTES, stream>>>(
        es, vs, ed, vd, cn_b1, cn_b2, bn_b1, bn_b2, wblob, lsh);
    llr_kernel<<<BNP / 256, 256, 0, stream>>>(ed, vd, llr_w, llr_b, loss,
                                              pred1, pred2, lvl);
    _Float16* te = (_Float16*)es;  es = ed;  ed = te;
    unsigned char* tv = (unsigned char*)vs;  vs = vd;  vd = tv;
  }
}
